// SpMiddleFHD_17025250362008
// MI455X (gfx1250) — compile-verified, matrix operand order checked by reading, not run on hardware
//
#include <hip/hip_runtime.h>
#include <cstdint>
#include <cstddef>

// ---------------------------------------------------------------------------
// CDNA5 (gfx1250) wave32 WMMA types
// ---------------------------------------------------------------------------
typedef __attribute__((ext_vector_type(16))) _Float16 v16h;
typedef __attribute__((ext_vector_type(8)))  float    v8f;
typedef __attribute__((ext_vector_type(4)))  float    f32x4;

union VU { v16h h; f32x4 f[2]; };

// ---------------------------------------------------------------------------
// Scatter voxel features into dense f16 grid (Cin padded to 16) + mask
// ---------------------------------------------------------------------------
__global__ void k_scatter(const float* __restrict__ vf, const int* __restrict__ coors,
                          _Float16* __restrict__ dense, float* __restrict__ mask,
                          int N, int D, int H, int W, int C) {
  int i = blockIdx.x * blockDim.x + threadIdx.x;
  if (i >= N) return;
  int d = coors[4*i+1], h = coors[4*i+2], w = coors[4*i+3];
  size_t p = ((size_t)d * H + h) * W + w;
#pragma unroll
  for (int c = 0; c < 4; ++c) dense[p * C + c] = (_Float16)vf[4*i + c];
  mask[p] = 1.0f;
}

// ---------------------------------------------------------------------------
// 3x3x3 stride-2 max-pool > 0 on mask
// ---------------------------------------------------------------------------
__global__ void k_maskpool(const float* __restrict__ in, float* __restrict__ out,
                           int iD, int iH, int iW, int oD, int oH, int oW) {
  int i = blockIdx.x * blockDim.x + threadIdx.x;
  int total = oD * oH * oW;
  if (i >= total) return;
  int x = i % oW, y = (i / oW) % oH, z = i / (oW * oH);
  float m = 0.f;
  for (int dz = 0; dz < 3; ++dz)
    for (int dy = 0; dy < 3; ++dy)
      for (int dx = 0; dx < 3; ++dx) {
        int iz = 2*z + dz - 1, iy = 2*y + dy - 1, ix = 2*x + dx - 1;
        if (iz >= 0 && iz < iD && iy >= 0 && iy < iH && ix >= 0 && ix < iW)
          if (in[((size_t)iz * iH + iy) * iW + ix] > 0.f) m = 1.f;
      }
  out[i] = m;
}

// ---------------------------------------------------------------------------
// Pack conv weights OIDHW(f32) -> [tap][Cout][CinPad](f16), zero-padding Cin
// ---------------------------------------------------------------------------
__global__ void k_pack(const float* __restrict__ w, _Float16* __restrict__ wpk,
                       int taps, int co, int ciIn, int ciPad) {
  int i = blockIdx.x * blockDim.x + threadIdx.x;
  int total = taps * co * ciPad;
  if (i >= total) return;
  int ch = i % ciPad;
  int o  = (i / ciPad) % co;
  int t  = i / (ciPad * co);
  float v = (ch < ciIn) ? w[((size_t)o * ciIn + ch) * taps + t] : 0.f;
  wpk[i] = (_Float16)v;
}

// ---------------------------------------------------------------------------
// Implicit-GEMM conv (3D or 2D) with V_WMMA_F32_16X16X32_F16.
// Fully templated so all address math constant-folds and the K loop fully
// unrolls into a stream of global_load_b128 + v_wmma instructions.
// Activations channel-last f16 [D][H][W][CIN]; weights [tap][COUT][CIN] f16.
// One wave computes a 16 (W-positions) x 16 (Cout) tile; WAVES waves cover
// WAVES*16 output channels; blockIdx.z = oz*CB + cb splits remaining Cout.
// Epilogue: f32 BN scale/bias + ReLU + mask; f16 out and/or f32 NCHW out.
// ---------------------------------------------------------------------------
template <int CIN, int COUT, int WAVES, int KD, int KH, int KW, int STRIDE, int PAD>
__global__ void k_conv_wmma(const _Float16* __restrict__ in,
                            const _Float16* __restrict__ wpk,
                            const float* __restrict__ scale,
                            const float* __restrict__ bias,
                            const float* __restrict__ mask,
                            _Float16* __restrict__ out,
                            float* __restrict__ out32,
                            int inD, int inH, int inW,
                            int outD, int outH, int outW) {
  constexpr int PD = (KD > 1) ? PAD : 0;
  constexpr int PH = (KH > 1) ? PAD : 0;
  constexpr int PW = (KW > 1) ? PAD : 0;
  constexpr int KHKW = KH * KW;
  constexpr int Ktot = KD * KHKW * CIN;
  constexpr int CPB  = WAVES * 16;
  constexpr int CB   = COUT / CPB;

  const int lane = threadIdx.x;
  const int hi   = lane >> 4;      // half-select per ISA 16-bit A layout
  const int lr   = lane & 15;      // A row (M) / B column (N) within tile
  const int cb   = (CB > 1) ? ((int)blockIdx.z % CB) : 0;
  const int oz   = (CB > 1) ? ((int)blockIdx.z / CB) : (int)blockIdx.z;
  const int oy   = blockIdx.y;
  const int w0   = blockIdx.x * 16;
  const int n    = cb * CPB + threadIdx.y * 16 + lr;

  v8f acc = {0.f, 0.f, 0.f, 0.f, 0.f, 0.f, 0.f, 0.f};

#pragma unroll
  for (int kk = 0; kk < Ktot; kk += 32) {
    VU a, b;
#pragma unroll
    for (int hh = 0; hh < 2; ++hh) {
      int g = kk + hh * 16 + hi * 8;     // 8-element group start (K index)
      f32x4 av = {0.f, 0.f, 0.f, 0.f};
      f32x4 bv = {0.f, 0.f, 0.f, 0.f};
      if (kk + hh * 16 + 8 < Ktot + 1) { // tail guard (folds; groups are 8-wide)
        int t  = g / CIN;                // CIN is a power of two -> shift
        int ch = g & (CIN - 1);
        int dz = t / KHKW;
        int r2 = t - dz * KHKW;
        int dy = r2 / KW;
        int dx = r2 - dy * KW;
        int iz = oz * STRIDE + dz - PD;
        int iy = oy * STRIDE + dy - PH;
        int ix = (w0 + lr) * STRIDE + dx - PW;
        bool okk = (g < Ktot);
        if (okk && iz >= 0 && iz < inD && iy >= 0 && iy < inH && ix >= 0 && ix < inW)
          av = *(const f32x4*)(in + (((size_t)iz * inH + iy) * inW + ix) * CIN + ch);
        if (okk)
          bv = *(const f32x4*)(wpk + ((size_t)t * COUT + n) * CIN + ch);
      }
      a.f[hh] = av;
      b.f[hh] = bv;
    }
    acc = __builtin_amdgcn_wmma_f32_16x16x32_f16(false, a.h, false, b.h,
                                                 (short)0, acc, false, false);
  }

  const float sc = scale[n];
  const float bi = bias[n];
  const size_t rowBase = ((size_t)oz * outH + oy) * outW;
#pragma unroll
  for (int r = 0; r < 8; ++r) {
    int mm = r + hi * 8;
    int ow = w0 + mm;
    if (ow >= outW) continue;
    float v = acc[r] * sc + bi;
    v = fmaxf(v, 0.f);
    if (mask) v *= mask[rowBase + ow];
    if (out)   out[(rowBase + ow) * COUT + n] = (_Float16)v;
    if (out32) out32[(size_t)n * outH * outW + (size_t)oy * outW + ow] = v; // NCHW (outD==1)
  }
}

template <int CIN, int COUT, int WAVES, int KD, int KH, int KW, int STRIDE, int PAD>
static void conv_launch(hipStream_t st, const _Float16* in, const _Float16* wpk,
                        const float* s, const float* b, const float* mask,
                        _Float16* out, float* out32,
                        int inD, int inH, int inW, int outD, int outH, int outW) {
  constexpr int CB = COUT / (WAVES * 16);
  dim3 grid((outW + 15) / 16, outH, outD * CB);
  dim3 block(32, WAVES);
  k_conv_wmma<CIN, COUT, WAVES, KD, KH, KW, STRIDE, PAD>
      <<<grid, block, 0, st>>>(in, wpk, s, b, mask, out, out32,
                               inD, inH, inW, outD, outH, outW);
}

// ---------------------------------------------------------------------------
// (1,64,2,50,50) channel-last -> BEV (50,50,128) channel-last, ch = c*2+z
// ---------------------------------------------------------------------------
__global__ void k_bev_repack(const _Float16* __restrict__ x3, _Float16* __restrict__ bev) {
  int i = blockIdx.x * blockDim.x + threadIdx.x;
  if (i >= 2 * 50 * 50 * 64) return;
  int c = i % 64;
  int w = (i / 64) % 50;
  int h = (i / 3200) % 50;
  int z = i / 160000;
  bev[((h * 50 + w) * 128) + c * 2 + z] = x3[i];
}

// ---------------------------------------------------------------------------
// points_mean = (batch, x, y, z)
// ---------------------------------------------------------------------------
__global__ void k_points_mean(const int* __restrict__ coors, const float* __restrict__ vf,
                              float* __restrict__ pm, int P) {
  int i = blockIdx.x * blockDim.x + threadIdx.x;
  if (i >= P) return;
  pm[4*i+0] = (float)coors[4*i+0];
  pm[4*i+1] = vf[4*i+0];
  pm[4*i+2] = vf[4*i+1];
  pm[4*i+3] = vf[4*i+2];
}

// ---------------------------------------------------------------------------
// Build known coords (kx) + gather known feats (kf, f16->f32) for one level
// ---------------------------------------------------------------------------
__global__ void k_build_known(const int* __restrict__ cl, int n,
                              const _Float16* __restrict__ mid, int C,
                              int Dm, int Hm, int Wm,
                              float vs0, float vs1, float vs2,
                              float* __restrict__ kx, float* __restrict__ kf) {
  int j = blockIdx.x * blockDim.x + threadIdx.x;
  if (j >= n) return;
  int b = cl[4*j+0], d = cl[4*j+1], h = cl[4*j+2], w = cl[4*j+3];
  kx[4*j+0] = (float)b;
  kx[4*j+1] = w * vs0 + 0.0f   + 0.5f * vs0;
  kx[4*j+2] = h * vs1 + -40.0f + 0.5f * vs1;
  kx[4*j+3] = d * vs2 + -3.0f  + 0.5f * vs2;
  const _Float16* src = mid + (((size_t)d * Hm + h) * Wm + w) * C;
  for (int c = 0; c < C; ++c) kf[(size_t)j * C + c] = (float)src[c];
}

// ---------------------------------------------------------------------------
// Brute-force 3-NN interpolation (LDS-tiled over known points)
// ---------------------------------------------------------------------------
#define ITILE 256
__global__ void k_interp(const float* __restrict__ pts, int P,
                         const float* __restrict__ kx, const float* __restrict__ kf,
                         int n, int C, float* __restrict__ feats, int FST, int coff) {
  __shared__ float sk[ITILE * 4];
  int p = blockIdx.x * blockDim.x + threadIdx.x;
  float pb = 0.f, px = 0.f, py = 0.f, pz = 0.f;
  if (p < P) { pb = pts[4*p]; px = pts[4*p+1]; py = pts[4*p+2]; pz = pts[4*p+3]; }
  float d0 = 3.0e38f, d1 = 3.0e38f, d2 = 3.0e38f;
  int i0 = 0, i1 = 0, i2 = 0;
  for (int base = 0; base < n; base += ITILE) {
    int cnt = (n - base < ITILE) ? (n - base) : ITILE;
    __syncthreads();
    for (int t = threadIdx.x; t < cnt * 4; t += blockDim.x) sk[t] = kx[base * 4 + t];
    __syncthreads();
    for (int j = 0; j < cnt; ++j) {
      float dx = px - sk[4*j+1];
      float dy = py - sk[4*j+2];
      float dz = pz - sk[4*j+3];
      float d  = dx*dx + dy*dy + dz*dz;
      d = (pb == sk[4*j]) ? d : 1.0e10f;
      int gi = base + j;
      if (d < d0)      { d2=d1; i2=i1; d1=d0; i1=i0; d0=d; i0=gi; }
      else if (d < d1) { d2=d1; i2=i1; d1=d;  i1=gi; }
      else if (d < d2) { d2=d;  i2=gi; }
    }
  }
  if (p >= P) return;
  float r0 = 1.f/(d0 + 1e-8f), r1 = 1.f/(d1 + 1e-8f), r2 = 1.f/(d2 + 1e-8f);
  float s = r0 + r1 + r2;
  r0 /= s; r1 /= s; r2 /= s;
  const float* f0 = kf + (size_t)i0 * C;
  const float* f1 = kf + (size_t)i1 * C;
  const float* f2 = kf + (size_t)i2 * C;
  float* o = feats + (size_t)p * FST + coff;
  for (int c = 0; c < C; ++c) o[c] = r0*f0[c] + r1*f1[c] + r2*f2[c];
}

// ---------------------------------------------------------------------------
// Point head: pointwise = feats(160) @ fc^T(64), cls = pw @ clsw^T, reg(3)
// FC weights staged in LDS (40KB), broadcast reads (conflict-free).
// ---------------------------------------------------------------------------
__global__ void k_point_head(const float* __restrict__ feats,
                             const float* __restrict__ fcw,   // [64][160]
                             const float* __restrict__ clsw,  // [64]
                             const float* __restrict__ regw,  // [3][64]
                             float* __restrict__ cls, float* __restrict__ reg, int P) {
  __shared__ float sw[64 * 160];
  for (int t = threadIdx.x; t < 64 * 160; t += blockDim.x) sw[t] = fcw[t];
  __syncthreads();
  int p = blockIdx.x * blockDim.x + threadIdx.x;
  if (p >= P) return;
  const float* f = feats + (size_t)p * 160;
  float pw[64];
#pragma unroll
  for (int o = 0; o < 64; ++o) pw[o] = 0.f;
  for (int k = 0; k < 160; ++k) {
    float fk = f[k];
#pragma unroll
    for (int o = 0; o < 64; ++o) pw[o] += fk * sw[o * 160 + k];
  }
  float c0 = 0.f;
#pragma unroll
  for (int o = 0; o < 64; ++o) c0 += pw[o] * clsw[o];
  cls[p] = c0;
  for (int r = 0; r < 3; ++r) {
    float rv = 0.f;
#pragma unroll
    for (int o = 0; o < 64; ++o) rv += pw[o] * regw[r * 64 + o];
    reg[p * 3 + r] = rv;
  }
}

// ---------------------------------------------------------------------------
// Host driver
// ---------------------------------------------------------------------------
extern "C" void kernel_launch(void* const* d_in, const int* in_sizes, int n_in,
                              void* d_out, int out_size, void* d_ws, size_t ws_size,
                              hipStream_t stream) {
  (void)n_in; (void)out_size; (void)ws_size;
  // Input layout: dict leaves in order.
  const float* vf = (const float*)d_in[0];
  const float* vxw[14]; const float* vxs[14]; const float* vxb[14];
  for (int i = 0; i < 14; ++i) {
    vxw[i] = (const float*)d_in[1 + i];
    vxs[i] = (const float*)d_in[15 + i];
    vxb[i] = (const float*)d_in[29 + i];
  }
  const float* bw[8]; const float* bs[8]; const float* bb[8];
  for (int i = 0; i < 8; ++i) {
    bw[i] = (const float*)d_in[43 + i];
    bs[i] = (const float*)d_in[51 + i];
    bb[i] = (const float*)d_in[59 + i];
  }
  const float* fcw  = (const float*)d_in[67];
  const float* clsw = (const float*)d_in[68];
  const float* regw = (const float*)d_in[69];
  const int* coors  = (const int*)d_in[70];
  const int* c1 = (const int*)d_in[71];
  const int* c2 = (const int*)d_in[72];
  const int* c3 = (const int*)d_in[73];
  const int n1 = in_sizes[71] / 4, n2 = in_sizes[72] / 4, n3 = in_sizes[73] / 4;
  const int P  = in_sizes[0] / 4;

  // bump allocator over workspace
  char* wsb = (char*)d_ws;
  size_t off = 0;
  auto alloc = [&](size_t bytes) -> void* {
    void* p = wsb + off;
    off += (bytes + 255) & ~(size_t)255;
    return p;
  };

  _Float16* dense0 = (_Float16*)alloc((size_t)16*400*400*16*2);
  _Float16* A0     = (_Float16*)alloc((size_t)16*400*400*16*2);
  _Float16* B0     = (_Float16*)alloc((size_t)16*400*400*16*2);
  _Float16* r1a    = (_Float16*)alloc((size_t)8*200*200*32*2);
  _Float16* r1b    = (_Float16*)alloc((size_t)8*200*200*32*2);
  _Float16* r2a    = (_Float16*)alloc((size_t)4*100*100*64*2);
  _Float16* r2b    = (_Float16*)alloc((size_t)4*100*100*64*2);
  _Float16* r3a    = (_Float16*)alloc((size_t)2*50*50*64*2);
  _Float16* r3b    = (_Float16*)alloc((size_t)2*50*50*64*2);
  float* mask0 = (float*)alloc((size_t)16*400*400*4);
  float* mask1 = (float*)alloc((size_t)8*200*200*4);
  float* mask2 = (float*)alloc((size_t)4*100*100*4);
  float* mask3 = (float*)alloc((size_t)2*50*50*4);
  _Float16* bev_in = (_Float16*)alloc((size_t)2500*128*2);
  _Float16* bev_a  = (_Float16*)alloc((size_t)2500*256*2);
  _Float16* bev_b  = (_Float16*)alloc((size_t)2500*256*2);
  _Float16* wpk    = (_Float16*)alloc((size_t)9*256*256*2);
  float* feats = (float*)alloc((size_t)P*160*4);
  float* kx1 = (float*)alloc((size_t)n1*4*4);
  float* kf1 = (float*)alloc((size_t)n1*32*4);
  float* kx2 = (float*)alloc((size_t)n2*4*4);
  float* kf2 = (float*)alloc((size_t)n2*64*4);
  float* kx3 = (float*)alloc((size_t)n3*4*4);
  float* kf3 = (float*)alloc((size_t)n3*64*4);

  // Output regions (all f32). y/conv6 are NCHW.
  float* yout     = (float*)d_out;
  float* conv6out = yout + 640000;
  float* pmout    = yout + 1280000;
  float* clsout   = yout + 1361920;
  float* regout   = yout + 1382400;

  // --- init + scatter + mask pyramid ---
  hipMemsetAsync(dense0, 0, (size_t)16*400*400*16*2, stream);
  hipMemsetAsync(mask0, 0, (size_t)16*400*400*4, stream);
  k_scatter<<<(P + 255) / 256, 256, 0, stream>>>(vf, coors, dense0, mask0, P, 16, 400, 400, 16);
  k_maskpool<<<(8*200*200 + 255) / 256, 256, 0, stream>>>(mask0, mask1, 16, 400, 400, 8, 200, 200);
  k_maskpool<<<(4*100*100 + 255) / 256, 256, 0, stream>>>(mask1, mask2, 8, 200, 200, 4, 100, 100);
  k_maskpool<<<(2*50*50 + 255) / 256, 256, 0, stream>>>(mask2, mask3, 4, 100, 100, 2, 50, 50);

  // --- 14 sparse-middle 3D conv layers (WMMA implicit GEMM) ---
  // weight pack helper
  auto pack = [&](const float* w, int taps, int co, int ciIn, int ciPad) {
    int tot = taps * co * ciPad;
    k_pack<<<(tot + 255) / 256, 256, 0, stream>>>(w, wpk, taps, co, ciIn, ciPad);
  };

  // L0: 4(pad16)->16, s1, res0
  pack(vxw[0], 27, 16, 4, 16);
  conv_launch<16,16,1,3,3,3,1,1>(stream, dense0, wpk, vxs[0], vxb[0], mask0, A0, nullptr,
                                 16,400,400, 16,400,400);
  // L1: 16->16, s1, res0
  pack(vxw[1], 27, 16, 16, 16);
  conv_launch<16,16,1,3,3,3,1,1>(stream, A0, wpk, vxs[1], vxb[1], mask0, B0, nullptr,
                                 16,400,400, 16,400,400);
  // L2: 16->32, s2, res0->res1
  pack(vxw[2], 27, 32, 16, 16);
  conv_launch<16,32,2,3,3,3,2,1>(stream, B0, wpk, vxs[2], vxb[2], mask1, r1a, nullptr,
                                 16,400,400, 8,200,200);
  // L3: 32->32, s1, res1
  pack(vxw[3], 27, 32, 32, 32);
  conv_launch<32,32,2,3,3,3,1,1>(stream, r1a, wpk, vxs[3], vxb[3], mask1, r1b, nullptr,
                                 8,200,200, 8,200,200);
  // L4: 32->32, s1, res1  (MID1 = r1a)
  pack(vxw[4], 27, 32, 32, 32);
  conv_launch<32,32,2,3,3,3,1,1>(stream, r1b, wpk, vxs[4], vxb[4], mask1, r1a, nullptr,
                                 8,200,200, 8,200,200);
  // L5: 32->64, s2, res1->res2
  pack(vxw[5], 27, 64, 32, 32);
  conv_launch<32,64,4,3,3,3,2,1>(stream, r1a, wpk, vxs[5], vxb[5], mask2, r2a, nullptr,
                                 8,200,200, 4,100,100);
  // L6..L8: 64->64, s1, res2  (MID2 = r2b after L8)
  pack(vxw[6], 27, 64, 64, 64);
  conv_launch<64,64,4,3,3,3,1,1>(stream, r2a, wpk, vxs[6], vxb[6], mask2, r2b, nullptr,
                                 4,100,100, 4,100,100);
  pack(vxw[7], 27, 64, 64, 64);
  conv_launch<64,64,4,3,3,3,1,1>(stream, r2b, wpk, vxs[7], vxb[7], mask2, r2a, nullptr,
                                 4,100,100, 4,100,100);
  pack(vxw[8], 27, 64, 64, 64);
  conv_launch<64,64,4,3,3,3,1,1>(stream, r2a, wpk, vxs[8], vxb[8], mask2, r2b, nullptr,
                                 4,100,100, 4,100,100);
  // L9: 64->64, s2, res2->res3
  pack(vxw[9], 27, 64, 64, 64);
  conv_launch<64,64,4,3,3,3,2,1>(stream, r2b, wpk, vxs[9], vxb[9], mask3, r3a, nullptr,
                                 4,100,100, 2,50,50);
  // L10..L12: 64->64, s1, res3  (MID3 = r3b after L12)
  pack(vxw[10], 27, 64, 64, 64);
  conv_launch<64,64,4,3,3,3,1,1>(stream, r3a, wpk, vxs[10], vxb[10], mask3, r3b, nullptr,
                                 2,50,50, 2,50,50);
  pack(vxw[11], 27, 64, 64, 64);
  conv_launch<64,64,4,3,3,3,1,1>(stream, r3b, wpk, vxs[11], vxb[11], mask3, r3a, nullptr,
                                 2,50,50, 2,50,50);
  pack(vxw[12], 27, 64, 64, 64);
  conv_launch<64,64,4,3,3,3,1,1>(stream, r3a, wpk, vxs[12], vxb[12], mask3, r3b, nullptr,
                                 2,50,50, 2,50,50);
  // L13: 'e' 1x1x1, 64->64, res3
  pack(vxw[13], 1, 64, 64, 64);
  conv_launch<64,64,4,1,1,1,1,0>(stream, r3b, wpk, vxs[13], vxb[13], mask3, r3a, nullptr,
                                 2,50,50, 2,50,50);

  // --- BEV stack ---
  k_bev_repack<<<(2*50*50*64 + 255) / 256, 256, 0, stream>>>(r3a, bev_in);
  // L0: 128->256 3x3
  pack(bw[0], 9, 256, 128, 128);
  conv_launch<128,256,4,1,3,3,1,1>(stream, bev_in, wpk, bs[0], bb[0], nullptr, bev_a, nullptr,
                                   1,50,50, 1,50,50);
  // L1..L6: 256->256 3x3 (ping-pong); L6 also writes conv6 (f32 NCHW)
  _Float16* bufs[2] = {bev_a, bev_b};
  const _Float16* bcur = bev_a;
  for (int i = 1; i < 7; ++i) {
    pack(bw[i], 9, 256, 256, 256);
    _Float16* bout = bufs[i & 1];
    float* o32 = (i == 6) ? conv6out : nullptr;
    conv_launch<256,256,4,1,3,3,1,1>(stream, bcur, wpk, bs[i], bb[i], nullptr, bout, o32,
                                     1,50,50, 1,50,50);
    bcur = bout;
  }
  // L7: 256->256 1x1 -> y (f32 NCHW only)
  pack(bw[7], 1, 256, 256, 256);
  conv_launch<256,256,4,1,1,1,1,0>(stream, bcur, wpk, bs[7], bb[7], nullptr, nullptr, yout,
                                   1,50,50, 1,50,50);

  // --- point branch ---
  k_points_mean<<<(P + 255) / 256, 256, 0, stream>>>(coors, vf, pmout, P);

  k_build_known<<<(n1 + 255) / 256, 256, 0, stream>>>(c1, n1, r1a, 32, 8, 200, 200,
                                                      0.1f, 0.1f, 0.2f, kx1, kf1);
  k_interp<<<(P + 255) / 256, 256, 0, stream>>>(pmout, P, kx1, kf1, n1, 32, feats, 160, 0);

  k_build_known<<<(n2 + 255) / 256, 256, 0, stream>>>(c2, n2, r2b, 64, 4, 100, 100,
                                                      0.2f, 0.2f, 0.4f, kx2, kf2);
  k_interp<<<(P + 255) / 256, 256, 0, stream>>>(pmout, P, kx2, kf2, n2, 64, feats, 160, 32);

  k_build_known<<<(n3 + 255) / 256, 256, 0, stream>>>(c3, n3, r3b, 64, 2, 50, 50,
                                                      0.4f, 0.4f, 0.8f, kx3, kf3);
  k_interp<<<(P + 255) / 256, 256, 0, stream>>>(pmout, P, kx3, kf3, n3, 64, feats, 160, 96);

  k_point_head<<<(P + 255) / 256, 256, 0, stream>>>(feats, fcw, clsw, regw, clsout, regout, P);
}
